// CLIPLoss_88244398064307
// MI455X (gfx1250) — matrix operand order repaired and checked by reading, not compiled, not run on hardware
//
#include <hip/hip_runtime.h>
#include <math.h>

typedef unsigned short u16;
typedef __attribute__((ext_vector_type(8)))  __bf16 v8bf;
typedef __attribute__((ext_vector_type(16))) __bf16 v16bf;
typedef __attribute__((ext_vector_type(8)))  float  v8f;

#define N_TOTAL 16384
#define DDIM 256
#define WAVES 4
#define ROWS_PER_BLOCK (WAVES * 16)   // 64 rows per block, 16 per wave
#define LDS_STRIDE 264                // bf16 elems: 528B rows -> conflict-free, 16B aligned
#define CSPLIT 4                      // column-dimension split (gridDim.y)
#define COLS_PER_SPLIT (N_TOTAL / CSPLIT)

// ---- fp32 <-> bf16 (round-to-nearest-even) helpers ----------------------
__device__ __forceinline__ u16 f2bf(float f) {
    const unsigned u = __float_as_uint(f);
    return (u16)((u + 0x7FFFu + ((u >> 16) & 1u)) >> 16);
}
__device__ __forceinline__ float bf2f(u16 h) {
    return __uint_as_float(((unsigned)h) << 16);
}

// -------------------------------------------------------------------------
// Split fp32 -> (hi, lo) bf16 pair: v = hi + lo + O(2^-16 * v).
// -------------------------------------------------------------------------
__global__ __launch_bounds__(256) void split_kernel(const float* __restrict__ src,
                                                    u16* __restrict__ hi,
                                                    u16* __restrict__ lo) {
    const int i = (blockIdx.x * blockDim.x + threadIdx.x) * 4;
    const float4 v = *(const float4*)(src + i);
    ushort4 h, l;
    h.x = f2bf(v.x); l.x = f2bf(v.x - bf2f(h.x));
    h.y = f2bf(v.y); l.y = f2bf(v.y - bf2f(h.y));
    h.z = f2bf(v.z); l.z = f2bf(v.z - bf2f(h.z));
    h.w = f2bf(v.w); l.w = f2bf(v.w - bf2f(h.w));
    *(ushort4*)(hi + i) = h;
    *(ushort4*)(lo + i) = l;
}

// ---- WMMA fragment loaders (16-bit layouts, CDNA5 ISA §7.12.2) ---------
// A 16x32: lane holds row l&15; K = [8g..8g+7] then [16+8g..16+8g+7], g=l>>4.
__device__ __forceinline__ v16bf load_a_frag(const u16* __restrict__ rowPtr,
                                             int kb, int g) {
    const v8bf a0 = *(const v8bf*)(rowPtr + kb + 8 * g);
    const v8bf a1 = *(const v8bf*)(rowPtr + kb + 16 + 8 * g);
    return __builtin_shufflevector(a0, a1, 0, 1, 2, 3, 4, 5, 6, 7,
                                           8, 9, 10, 11, 12, 13, 14, 15);
}
// B 32x16: lane holds col l&15; K = [8g..8g+7] then [16+8g..16+8g+7], g=l>>4.
__device__ __forceinline__ v16bf load_b_frag(const u16* __restrict__ rowPtr,
                                             int kb, int g) {
    const v8bf b0 = *(const v8bf*)(rowPtr + kb + 8 * g);
    const v8bf b1 = *(const v8bf*)(rowPtr + kb + 16 + 8 * g);
    return __builtin_shufflevector(b0, b1, 0, 1, 2, 3, 4, 5, 6, 7,
                                           8, 9, 10, 11, 12, 13, 14, 15);
}

// -------------------------------------------------------------------------
// Fused GEMM + online row-logsumexp over a column slice, bf16 split-product:
//   logits ~= hiA·hiB + hiA·loB + loA·hiB   (fp32 accumulate, ~1e-4 rel err)
// Each CTA: 64 rows vs a 4096-column slice; one wave owns 16 rows; columns
// stream 16 at a time; 3 independent v_wmma_f32_16x16x32_bf16 chains.
// Emits per-row partial (max, sumexp); merged exactly in finalize.
// -------------------------------------------------------------------------
__global__ __launch_bounds__(128) void rowlse_kernel(const u16* __restrict__ Ahi,
                                                     const u16* __restrict__ Alo,
                                                     const u16* __restrict__ Bhi,
                                                     const u16* __restrict__ Blo,
                                                     float* __restrict__ mPart,
                                                     float* __restrict__ sPart) {
    extern __shared__ u16 smem[];                 // 2 * 64 * LDS_STRIDE bf16
    u16* sAhi = smem;
    u16* sAlo = smem + ROWS_PER_BLOCK * LDS_STRIDE;

    const int tid  = threadIdx.x;
    const int wave = tid >> 5;
    const int lane = tid & 31;
    const int n16  = lane & 15;
    const int g    = lane >> 4;       // half-wave: K-group select; C rows g*8+r
    const int rowBase = blockIdx.x * ROWS_PER_BLOCK;
    const int colBase = blockIdx.y * COLS_PER_SPLIT;

    // Stage this block's 64x256 hi/lo bf16 panels of A into LDS (16B chunks).
    for (int i = tid; i < ROWS_PER_BLOCK * (DDIM / 8); i += blockDim.x) {
        const int row = i >> 5;                   // 32 chunks of 8 elems per row
        const int c8  = (i & 31) * 8;
        const size_t gof = (size_t)(rowBase + row) * DDIM + c8;
        *(v8bf*)&sAhi[row * LDS_STRIDE + c8] = *(const v8bf*)(Ahi + gof);
        *(v8bf*)&sAlo[row * LDS_STRIDE + c8] = *(const v8bf*)(Alo + gof);
    }
    __syncthreads();

    const u16* aHiRow = &sAhi[(wave * 16 + n16) * LDS_STRIDE];
    const u16* aLoRow = &sAlo[(wave * 16 + n16) * LDS_STRIDE];

    float mx[8], sm[8];
#pragma unroll
    for (int r = 0; r < 8; ++r) { mx[r] = -INFINITY; sm[r] = 0.0f; }

    for (int nt = 0; nt < COLS_PER_SPLIT / 16; ++nt) {
        const size_t bRow = (size_t)(colBase + nt * 16 + n16) * DDIM;
        const u16* bHiRow = Bhi + bRow;
        const u16* bLoRow = Blo + bRow;
        if (nt + 1 < COLS_PER_SPLIT / 16) {      // global_prefetch_b8 hints
            __builtin_prefetch(bHiRow + (size_t)DDIM * 16, 0, 1);
            __builtin_prefetch(bLoRow + (size_t)DDIM * 16, 0, 1);
        }

        v8f c0 = {}, c1 = {}, c2 = {};           // three independent WMMA chains
#pragma unroll
        for (int ks = 0; ks < DDIM / 32; ++ks) {
            const int kb = ks * 32;
            const v16bf aH = load_a_frag(aHiRow, kb, g);
            const v16bf aL = load_a_frag(aLoRow, kb, g);
            const v16bf bH = load_b_frag(bHiRow, kb, g);
            const v16bf bL = load_b_frag(bLoRow, kb, g);
            c0 = __builtin_amdgcn_wmma_f32_16x16x32_bf16(
                    false, aH, false, bH, (short)0, c0, false, false);
            c1 = __builtin_amdgcn_wmma_f32_16x16x32_bf16(
                    false, aH, false, bL, (short)0, c1, false, false);
            c2 = __builtin_amdgcn_wmma_f32_16x16x32_bf16(
                    false, aL, false, bH, (short)0, c2, false, false);
        }

        // Online logsumexp update: lane half g holds rows g*8+r, col n16.
#pragma unroll
        for (int r = 0; r < 8; ++r) {
            const float v  = c0[r] + (c1[r] + c2[r]);
            const float nm = fmaxf(mx[r], v);
            sm[r] = sm[r] * __expf(mx[r] - nm) + __expf(v - nm);
            mx[r] = nm;
        }
    }

    // Merge the 16 per-column partial (max,sum) pairs within each half-wave.
#pragma unroll
    for (int r = 0; r < 8; ++r) {
        float m = mx[r], s = sm[r];
#pragma unroll
        for (int off = 1; off < 16; off <<= 1) {
            const float mo = __shfl_xor(m, off, 32);
            const float so = __shfl_xor(s, off, 32);
            const float nm = fmaxf(m, mo);
            s = s * __expf(m - nm) + so * __expf(mo - nm);
            m = nm;
        }
        mx[r] = m; sm[r] = s;
    }

    if (n16 == 0) {
        const int rbase = rowBase + wave * 16 + g * 8;
        float* mOut = mPart + (size_t)blockIdx.y * N_TOTAL;
        float* sOut = sPart + (size_t)blockIdx.y * N_TOTAL;
#pragma unroll
        for (int r = 0; r < 8; ++r) {
            mOut[rbase + r] = mx[r];
            sOut[rbase + r] = sm[r];
        }
    }
}

// -------------------------------------------------------------------------
// diag[i] = <x_i, y_i> in exact fp32 — one wave per row.
// -------------------------------------------------------------------------
__global__ __launch_bounds__(256) void diag_kernel(const float* __restrict__ x,
                                                   const float* __restrict__ y,
                                                   float* __restrict__ d) {
    const int warp = threadIdx.x >> 5;
    const int lane = threadIdx.x & 31;
    const int row  = blockIdx.x * 8 + warp;
    const float* xr = x + (size_t)row * DDIM + lane * 8;
    const float* yr = y + (size_t)row * DDIM + lane * 8;

    float acc = 0.0f;
#pragma unroll
    for (int k = 0; k < 8; k += 4) {
        const float4 a = *(const float4*)(xr + k);
        const float4 b = *(const float4*)(yr + k);
        acc = fmaf(a.x, b.x, acc);
        acc = fmaf(a.y, b.y, acc);
        acc = fmaf(a.z, b.z, acc);
        acc = fmaf(a.w, b.w, acc);
    }
#pragma unroll
    for (int off = 16; off; off >>= 1) acc += __shfl_xor(acc, off, 32);
    if (lane == 0) d[row] = acc;
}

__device__ __forceinline__ float merge_lse(const float* __restrict__ mP,
                                           const float* __restrict__ sP, int i) {
    float m = -INFINITY;
#pragma unroll
    for (int p = 0; p < CSPLIT; ++p)
        m = fmaxf(m, mP[(size_t)p * N_TOTAL + i]);
    float s = 0.0f;
#pragma unroll
    for (int p = 0; p < CSPLIT; ++p)
        s += sP[(size_t)p * N_TOTAL + i] * __expf(mP[(size_t)p * N_TOTAL + i] - m);
    return m + __logf(s);
}

// -------------------------------------------------------------------------
// Merge column-split partials into per-row LSEs, then
// loss = 0.5 * ( mean_i(lse_row - diag) + mean_j(lse_col - diag) )
// -------------------------------------------------------------------------
__global__ __launch_bounds__(256) void finalize_kernel(const float* __restrict__ mR,
                                                       const float* __restrict__ sR,
                                                       const float* __restrict__ mC,
                                                       const float* __restrict__ sC,
                                                       const float* __restrict__ d,
                                                       float* __restrict__ out) {
    __shared__ float red[256];
    float acc = 0.0f;
    for (int i = threadIdx.x; i < N_TOTAL; i += 256) {
        const float lr = merge_lse(mR, sR, i);
        const float lc = merge_lse(mC, sC, i);
        acc += (lr - d[i]) + (lc - d[i]);
    }
    red[threadIdx.x] = acc;
    __syncthreads();
    for (int s = 128; s > 0; s >>= 1) {
        if (threadIdx.x < s) red[threadIdx.x] += red[threadIdx.x + s];
        __syncthreads();
    }
    if (threadIdx.x == 0) out[0] = 0.5f * red[0] / (float)N_TOTAL;
}

extern "C" void kernel_launch(void* const* d_in, const int* in_sizes, int n_in,
                              void* d_out, int out_size, void* d_ws, size_t ws_size,
                              hipStream_t stream) {
    const float* x = (const float*)d_in[0];
    const float* y = (const float*)d_in[1];
    float* out = (float*)d_out;

    const size_t ND = (size_t)N_TOTAL * DDIM;
    u16* xhi = (u16*)d_ws;             // bf16 arrays first: 16B+ aligned
    u16* xlo = xhi + ND;
    u16* yhi = xlo + ND;
    u16* ylo = yhi + ND;
    float* mR = (float*)(ylo + ND);    // CSPLIT * N_TOTAL each
    float* sR = mR + CSPLIT * N_TOTAL;
    float* mC = sR + CSPLIT * N_TOTAL;
    float* sC = mC + CSPLIT * N_TOTAL;
    float* dg = sC + CSPLIT * N_TOTAL; // N_TOTAL   (total ~35 MB of ws)

    // Precompute hi/lo bf16 splits of both inputs.
    split_kernel<<<ND / (256 * 4), 256, 0, stream>>>(x, xhi, xlo);
    split_kernel<<<ND / (256 * 4), 256, 0, stream>>>(y, yhi, ylo);

    const size_t shBytes = 2ull * ROWS_PER_BLOCK * LDS_STRIDE * sizeof(u16);
    const dim3 grid(N_TOTAL / ROWS_PER_BLOCK, CSPLIT);

    // image->text direction: row-LSE of x·yT
    rowlse_kernel<<<grid, 128, shBytes, stream>>>(xhi, xlo, yhi, ylo, mR, sR);
    // text->image direction: column-LSE of x·yT == row-LSE of y·xT
    rowlse_kernel<<<grid, 128, shBytes, stream>>>(yhi, ylo, xhi, xlo, mC, sC);

    diag_kernel<<<N_TOTAL / 8, 256, 0, stream>>>(x, y, dg);
    finalize_kernel<<<1, 256, 0, stream>>>(mR, sR, mC, sC, dg, out);
}
